// MambaBlock_30880814858914
// MI455X (gfx1250) — matrix operand order repaired and checked by reading, not compiled, not run on hardware
//
#include <hip/hip_runtime.h>
#include <hip/hip_bf16.h>
#include <math.h>

typedef __bf16 bf16;
typedef __attribute__((ext_vector_type(16))) __bf16 v16bf;
typedef __attribute__((ext_vector_type(8)))  __bf16 v8bf;
typedef __attribute__((ext_vector_type(8)))  float   v8f;

#define D_MODEL 1024
#define D_INNER 2048
#define D_STATE 16
#define DT_RANK 64
#define BATCH   4
#define SEQ     4096
#define ROWS    (BATCH * SEQ)           // 16384
#define XDBL_N  (DT_RANK + 2 * D_STATE) // 96

static __device__ __forceinline__ v16bf cat8(v8bf lo, v8bf hi) {
  return __builtin_shufflevector(lo, hi, 0,1,2,3,4,5,6,7,8,9,10,11,12,13,14,15);
}

static __device__ __forceinline__ uint32_t lds_addr(const void* p) {
  // generic (flat) pointer to LDS: low 32 bits are the wave-relative LDS byte
  // address (ISA 10.2 aperture mapping)
  return (uint32_t)(uintptr_t)p;
}

// issue one 16-bf16 (32B) half-row chunk: two async 16B copies, offset applies to
// both LDS and global addresses (ISA 8.2 pseudocode)
static __device__ __forceinline__ void async_copy_32bf16(uint32_t lds, const bf16* g) {
  asm volatile("global_load_async_to_lds_b128 %0, %1, off"
               :: "v"(lds), "v"(g) : "memory");
  asm volatile("global_load_async_to_lds_b128 %0, %1, off offset:16"
               :: "v"(lds), "v"(g) : "memory");
}

// ---------------- conversion / prep kernels ----------------

__global__ void k_cvt_bf16(const float* __restrict__ in, bf16* __restrict__ out, int n) {
  int i = blockIdx.x * 256 + threadIdx.x;
  if (i < n) out[i] = (bf16)in[i];
}

// in: f32 [K][N] row-major -> out: bf16 [N][K] (transposed)
__global__ void k_cvt_transpose(const float* __restrict__ in, bf16* __restrict__ out,
                                int K, int N) {
  int i = blockIdx.x * 256 + threadIdx.x;
  if (i < K * N) {
    int k = i / N, n = i - k * N;
    out[(size_t)n * K + k] = (bf16)in[i];
  }
}

__global__ void k_prep_A(const float* __restrict__ Alog, float* __restrict__ Aneg, int n) {
  int i = blockIdx.x * 256 + threadIdx.x;
  if (i < n) Aneg[i] = -__expf(Alog[i]);
}

// ---------------- bf16 WMMA GEMM ----------------
// C[M][N] = A[M][K] * B[K][N], B given pre-transposed as Bt[N][K].
// Block tile 128x128 (256 thr = 8 waves), wave tile 64x32, K-step 32.
// AMODE 0: A bf16, async double-buffered LDS staging (GLOBAL_LOAD_ASYNC_TO_LDS_B128)
// AMODE 1: A f32 (converted on stage-in), synchronous staging
// EPI: 0 = store bf16, 1 = store f32, 2 = bias + softplus, store bf16
// NGUARD: clamp/guard partial N tile (only GEMM2, N=96)
template<int AMODE, int EPI, bool NGUARD>
__global__ __launch_bounds__(256)
void k_gemm(const void* __restrict__ Aptr, int lda,
            const bf16* __restrict__ Bt,
            int M, int N, int K,
            void* __restrict__ Cptr, int ldc,
            const float* __restrict__ bias) {
  __shared__ __align__(16) bf16 As[2][128][40];
  __shared__ __align__(16) bf16 Bs[2][128][40];

  const int tid  = threadIdx.x;
  const int lane = tid & 31;
  const int wave = tid >> 5;
  const int m0 = blockIdx.y * 128;
  const int n0 = blockIdx.x * 128;
  const int wm = (wave >> 2) * 64;   // wave sub-tile row (2 wave-rows)
  const int wn = (wave & 3) * 32;    // wave sub-tile col (4 wave-cols)

  v8f acc[4][2] = {};

  const int lrow  = tid >> 1;        // 0..127: tile row (A) / tile col (Bt)
  const int lhalf = (tid & 1) * 16;  // 0 or 16: K chunk

  const int ar  = lane & 15;
  const int kb  = (lane >> 4) << 3;  // A frag K-interleave {0..7,16..23}/{8..15,24..31}
  const int kb2 = (lane >> 4) << 4;  // unused: B frag uses same K-interleave as A (kb)

  const int nk = K / 32;

  const bf16* aRow = (AMODE == 0)
      ? ((const bf16*)Aptr + (size_t)(m0 + lrow) * lda + lhalf) : (const bf16*)nullptr;
  int nIdx = n0 + lrow;
  if (NGUARD) nIdx = (nIdx < N) ? nIdx : (N - 1);   // clamp: branch-free edge tile
  const bf16* bRow = Bt + (size_t)nIdx * K + lhalf;

  if (AMODE == 0) {  // prologue: tile 0 -> buffer 0
    async_copy_32bf16(lds_addr(&As[0][lrow][lhalf]), aRow);
    async_copy_32bf16(lds_addr(&Bs[0][lrow][lhalf]), bRow);
  }

  for (int kk = 0; kk < nk; ++kk) {
    const int p = kk & 1;
    if (AMODE == 0) {
      if (kk + 1 < nk) {
        async_copy_32bf16(lds_addr(&As[1 - p][lrow][lhalf]), aRow + (kk + 1) * 32);
        async_copy_32bf16(lds_addr(&Bs[1 - p][lrow][lhalf]), bRow + (kk + 1) * 32);
        // 4 newer ops in flight; in-order completion => tile kk's 4 are done
        asm volatile("s_wait_asynccnt 0x4" ::: "memory");
      } else {
        asm volatile("s_wait_asynccnt 0x0" ::: "memory");
      }
      __syncthreads();
    } else {
      const float* srcA = (const float*)Aptr + (size_t)(m0 + lrow) * lda + kk * 32 + lhalf;
      #pragma unroll
      for (int j = 0; j < 16; ++j) As[p][lrow][lhalf + j] = (bf16)srcA[j];
      const bf16* srcB = bRow + kk * 32;
      *(v8bf*)&Bs[p][lrow][lhalf]     = *(const v8bf*)srcB;
      *(v8bf*)&Bs[p][lrow][lhalf + 8] = *(const v8bf*)(srcB + 8);
      __syncthreads();
    }

    // fragments + 8x v_wmma_f32_16x16x32_bf16
    v16bf af[4], bfg[2];
    #pragma unroll
    for (int mi = 0; mi < 4; ++mi) {
      const bf16* pA = &As[p][wm + mi * 16 + ar][0];
      af[mi] = cat8(*(const v8bf*)(pA + kb), *(const v8bf*)(pA + kb + 16));
    }
    #pragma unroll
    for (int ni = 0; ni < 2; ++ni) {
      const bf16* pB = &Bs[p][wn + ni * 16 + ar][0];
      bfg[ni] = cat8(*(const v8bf*)(pB + kb), *(const v8bf*)(pB + kb + 16));
    }
    #pragma unroll
    for (int mi = 0; mi < 4; ++mi)
      #pragma unroll
      for (int ni = 0; ni < 2; ++ni)
        acc[mi][ni] = __builtin_amdgcn_wmma_f32_16x16x32_bf16(
            false, af[mi], false, bfg[ni], (short)0, acc[mi][ni], false, false);

    __syncthreads();
  }

  // epilogue: C/D layout: VGPR i -> (M = i + 8*(lane>=16), N = lane&15)
  const int cBase = n0 + wn + (lane & 15);
  const int rBase = m0 + wm + ((lane >> 4) << 3);
  #pragma unroll
  for (int mi = 0; mi < 4; ++mi) {
    #pragma unroll
    for (int ni = 0; ni < 2; ++ni) {
      const int col = cBase + ni * 16;
      if (!NGUARD || col < N) {
        #pragma unroll
        for (int i = 0; i < 8; ++i) {
          const int row = rBase + mi * 16 + i;
          float v = acc[mi][ni][i];
          if (EPI == 2) {
            v += bias[col];
            v = (v > 20.f) ? v : log1pf(__expf(v));   // softplus
          }
          if (EPI == 1)
            ((float*)Cptr)[(size_t)row * ldc + col] = v;
          else
            ((bf16*)Cptr)[(size_t)row * ldc + col] = (bf16)v;
        }
      }
    }
  }
}

// ---------------- causal depthwise conv (k=4) + SiLU ----------------
__global__ __launch_bounds__(256)
void k_conv_silu(const bf16* __restrict__ xires, const float* __restrict__ w,
                 const float* __restrict__ bconv, bf16* __restrict__ xs) {
  int idx = blockIdx.x * 256 + threadIdx.x;   // ROWS * D_INNER threads
  int d = idx & (D_INNER - 1);
  int t = (idx >> 11) & (SEQ - 1);
  int b = idx >> 23;
  size_t rb = (size_t)b * SEQ;
  float acc = bconv[d];
  #pragma unroll
  for (int k = 0; k < 4; ++k) {
    int tt = t - 3 + k;
    if (tt >= 0) acc += w[d * 4 + k] * (float)xires[(rb + tt) * (2 * D_INNER) + d];
  }
  float s = acc / (1.f + __expf(-acc));
  xs[((size_t)(rb + t)) * D_INNER + d] = (bf16)s;
}

// ---------------- selective scan, fused skip + gate ----------------
__global__ __launch_bounds__(256)
void k_scan(const bf16* __restrict__ delta, const bf16* __restrict__ xs,
            const bf16* __restrict__ xires, const float* __restrict__ xdbl,
            const float* __restrict__ Aneg, const float* __restrict__ Dp,
            bf16* __restrict__ y) {
  int gid = blockIdx.x * 256 + threadIdx.x;   // 0 .. BATCH*D_INNER-1
  int b = gid >> 11;
  int d = gid & (D_INNER - 1);
  float Av[D_STATE];
  #pragma unroll
  for (int n = 0; n < D_STATE; ++n) Av[n] = Aneg[d * D_STATE + n];
  const float Dv = Dp[d];
  float h[D_STATE] = {};
  for (int t = 0; t < SEQ; ++t) {
    size_t row = (size_t)b * SEQ + t;
    float dt = (float)delta[row * D_INNER + d];
    float xv = (float)xs[row * D_INNER + d];
    float rv = (float)xires[row * (2 * D_INNER) + D_INNER + d];
    const float* bc = xdbl + row * XDBL_N;
    float du = dt * xv;
    float yacc = 0.f;
    #pragma unroll
    for (int n = 0; n < D_STATE; ++n) {
      float a = __expf(dt * Av[n]);
      h[n] = h[n] * a + du * bc[DT_RANK + n];
      yacc += h[n] * bc[DT_RANK + D_STATE + n];
    }
    float out = (yacc + xv * Dv) * (rv / (1.f + __expf(-rv)));
    y[row * D_INNER + d] = (bf16)out;
  }
}

// ---------------- host launcher ----------------

extern "C" void kernel_launch(void* const* d_in, const int* in_sizes, int n_in,
                              void* d_out, int out_size, void* d_ws, size_t ws_size,
                              hipStream_t stream) {
  const float* x      = (const float*)d_in[0];
  const float* W_in   = (const float*)d_in[1];
  const float* conv_w = (const float*)d_in[2];
  const float* conv_b = (const float*)d_in[3];
  const float* W_x    = (const float*)d_in[4];
  const float* W_dt   = (const float*)d_in[5];
  const float* b_dt   = (const float*)d_in[6];
  const float* W_out  = (const float*)d_in[7];
  const float* A_log  = (const float*)d_in[8];
  const float* Dvec   = (const float*)d_in[9];
  float* out = (float*)d_out;
  char* ws = (char*)d_ws;

  size_t off = 0;
  bf16* XB    = (bf16*)(ws + off); off += (size_t)ROWS * D_MODEL * 2;
  bf16* WINT  = (bf16*)(ws + off); off += (size_t)(2*D_INNER) * D_MODEL * 2;
  bf16* WXT   = (bf16*)(ws + off); off += (size_t)XDBL_N * D_INNER * 2;
  bf16* WDTT  = (bf16*)(ws + off); off += (size_t)D_INNER * DT_RANK * 2;
  bf16* WOUTT = (bf16*)(ws + off); off += (size_t)D_MODEL * D_INNER * 2;
  float* ANEG = (float*)(ws + off); off += (size_t)D_INNER * D_STATE * 4;
  bf16* XIRES = (bf16*)(ws + off); off += (size_t)ROWS * (2*D_INNER) * 2;
  bf16* XS    = (bf16*)(ws + off); off += (size_t)ROWS * D_INNER * 2;
  float* XDBL = (float*)(ws + off); off += (size_t)ROWS * XDBL_N * 4;
  bf16* DELTA = (bf16*)(ws + off); off += (size_t)ROWS * D_INNER * 2;
  bf16* Y     = (bf16*)(ws + off); off += (size_t)ROWS * D_INNER * 2;

  { int n = ROWS * D_MODEL;
    k_cvt_bf16<<<(n + 255) / 256, 256, 0, stream>>>(x, XB, n); }
  { int n = D_MODEL * (2*D_INNER);
    k_cvt_transpose<<<(n + 255) / 256, 256, 0, stream>>>(W_in, WINT, D_MODEL, 2*D_INNER); }
  { int n = D_INNER * XDBL_N;
    k_cvt_transpose<<<(n + 255) / 256, 256, 0, stream>>>(W_x, WXT, D_INNER, XDBL_N); }
  { int n = DT_RANK * D_INNER;
    k_cvt_transpose<<<(n + 255) / 256, 256, 0, stream>>>(W_dt, WDTT, DT_RANK, D_INNER); }
  { int n = D_INNER * D_MODEL;
    k_cvt_transpose<<<(n + 255) / 256, 256, 0, stream>>>(W_out, WOUTT, D_INNER, D_MODEL); }
  { int n = D_INNER * D_STATE;
    k_prep_A<<<(n + 255) / 256, 256, 0, stream>>>(A_log, ANEG, n); }

  // GEMM1: x @ W_in -> xi|res (bf16)
  k_gemm<0, 0, false><<<dim3((2*D_INNER)/128, ROWS/128), 256, 0, stream>>>(
      XB, D_MODEL, WINT, ROWS, 2*D_INNER, D_MODEL, XIRES, 2*D_INNER, nullptr);

  // causal depthwise conv + SiLU -> xs
  k_conv_silu<<<(ROWS * D_INNER) / 256, 256, 0, stream>>>(XIRES, conv_w, conv_b, XS);

  // GEMM2: xs @ W_x -> x_dbl (f32, N=96 edge-guarded)
  k_gemm<0, 1, true><<<dim3(1, ROWS/128), 256, 0, stream>>>(
      XS, D_INNER, WXT, ROWS, XDBL_N, D_INNER, XDBL, XDBL_N, nullptr);

  // GEMM3: softplus(dlt @ W_dt + b_dt) -> delta (A = f32 view of x_dbl[:, :64])
  k_gemm<1, 2, false><<<dim3(D_INNER/128, ROWS/128), 256, 0, stream>>>(
      XDBL, XDBL_N, WDTT, ROWS, D_INNER, DT_RANK, DELTA, D_INNER, b_dt);

  // selective scan + skip(D) + silu(res) gate -> y (bf16)
  k_scan<<<(BATCH * D_INNER) / 256, 256, 0, stream>>>(
      DELTA, XS, XIRES, XDBL, ANEG, Dvec, Y);

  // GEMM4: y @ W_out -> d_out (f32)
  k_gemm<0, 1, false><<<dim3(D_MODEL/128, ROWS/128), 256, 0, stream>>>(
      Y, D_INNER, WOUTT, ROWS, D_MODEL, D_INNER, out, D_MODEL, nullptr);

  (void)in_sizes; (void)n_in; (void)out_size; (void)ws_size;
}
